// ImprovedGNNModel_6648609374954
// MI455X (gfx1250) — compile-verified, matrix operand order checked by reading, not run on hardware
//
#include <hip/hip_runtime.h>

// ---------------------------------------------------------------------------
// GIN forward on MI455X (gfx1250).
// - GEMMs: bf16 WMMA (v_wmma_f32_16x16x32_bf16), f32 accumulate, fused BN/ReLU.
// - Weights are pre-swizzled in global memory into the native 16-bit B-fragment
//   layout; each GEMM workgroup stages them into LDS with a single TDM
//   tensor_load_to_lds (contiguous 32-48KB block), waited with s_wait_tensorcnt.
//   Bsw's address is escaped into a may-write asm so the optimizer cannot
//   treat the TDM-written LDS as store-free (loads would fold to undef).
// - Edge aggregation: wave-per-edge float4 gather + global f32 atomics
//   (HBM-bound: ~5 GB random traffic -> ~0.2 ms @ 23.3 TB/s; GEMMs ~24 GFLOP).
// ---------------------------------------------------------------------------

typedef __attribute__((ext_vector_type(16))) __bf16        v16bf;
typedef __attribute__((ext_vector_type(8)))  float         v8f;
typedef __attribute__((ext_vector_type(4)))  unsigned int  v4u;
typedef __attribute__((ext_vector_type(8)))  int           v8i;
typedef __attribute__((ext_vector_type(4)))  int           v4i;

#define H_DIM 128
#define F_IN 165
#define F_IN_PAD 192   // pad K to multiple of 32 for the WMMA K-loop
#define H2 64
#define NCLS 2

#if __has_builtin(__builtin_amdgcn_tensor_load_to_lds)
#define HAVE_TDM 1
#else
#define HAVE_TDM 0
#endif

__device__ __forceinline__ unsigned short f2bf(float f) {
    unsigned int u = __builtin_bit_cast(unsigned int, f);
    unsigned int r = u + 0x7fffu + ((u >> 16) & 1u);   // round-to-nearest-even
    return (unsigned short)(r >> 16);
}

// ----------------------------- fill ---------------------------------------
__global__ void fill_f32_kernel(float* __restrict__ p, long n, float v) {
    long stride = (long)gridDim.x * blockDim.x;
    for (long i = (long)blockIdx.x * blockDim.x + threadIdx.x; i < n; i += stride)
        p[i] = v;
}

// -------------------- x: f32 -> bf16 with column padding --------------------
__global__ void pad_cols_kernel(const float* __restrict__ src,
                                unsigned short* __restrict__ dst,
                                long R, int C, int Cpad) {
    long total = R * (long)Cpad;
    long stride = (long)gridDim.x * blockDim.x;
    for (long i = (long)blockIdx.x * blockDim.x + threadIdx.x; i < total; i += stride) {
        long r = i / Cpad;
        int  c = (int)(i - r * Cpad);
        dst[i] = (c < C) ? f2bf(src[r * C + c]) : (unsigned short)0;
    }
}

// ---------------- weight swizzle: f32 [K x 128] -> B-fragment order ---------
// Entry (kt, j, ln, vi) holds packed {W[k+1][n], W[k][n]} with
//   k = 32kt + 16(vi>>2) + 8(ln>>4) + 2(vi&3),  n = 16j + (ln&15)
// so in the GEMM each lane's 8-dword fragment is contiguous in LDS.
__global__ void swizzleW_kernel(const float* __restrict__ W,
                                unsigned int* __restrict__ wsw,
                                int K, int ktiles) {
    int total = ktiles * 2048;
    int stride = gridDim.x * blockDim.x;
    for (int idx = blockIdx.x * blockDim.x + threadIdx.x; idx < total; idx += stride) {
        int vi = idx & 7;
        int ln = (idx >> 3) & 31;
        int j  = (idx >> 8) & 7;
        int kt = idx >> 11;
        int k  = (kt << 5) + ((vi >> 2) << 4) + ((ln >> 4) << 3) + ((vi & 3) << 1);
        int n  = (j << 4) + (ln & 15);
        unsigned int lo = (k     < K) ? (unsigned int)f2bf(W[(long)k * H_DIM + n])       : 0u;
        unsigned int hi = (k + 1 < K) ? (unsigned int)f2bf(W[(long)(k + 1) * H_DIM + n]) : 0u;
        wsw[idx] = lo | (hi << 16);
    }
}

// ------------------------------- GEMM --------------------------------------
// C[M x 128] = act( bn( A[M x Kpad](bf16) @ W[Kpad x 128](bf16) + bias ) ) (+res)
// bn(v) = v*gamma + beta  ->  fused: val = acc*g + (bias*g + beta)
// Block: 256 threads = 8 waves; wave w owns rows [blk*128 + 16w, +16) x 128 cols.
// B (pre-swizzled in global) is staged to LDS via TDM tensor_load_to_lds.
__global__ __launch_bounds__(256) void gemm_bf16_wmma_kernel(
    const unsigned short* __restrict__ A,     // M x Kpad bf16
    const unsigned int*   __restrict__ Wsw,   // ktiles*2048 dwords, fragment order
    long M, int Kpad,
    const float* __restrict__ gamma,
    const float* __restrict__ bias,
    const float* __restrict__ beta,
    float* __restrict__ outF,                 // may be null
    unsigned short* __restrict__ outB,        // may be null
    const float* __restrict__ residual,       // may be null
    int do_relu)
{
    __shared__ unsigned int Bsw[6 * 8 * 32 * 8];   // up to K=192: 48 KB

    const int tid  = threadIdx.x;
    const int lane = tid & 31;
    const int wave = tid >> 5;
    const int ktiles = Kpad >> 5;

#if HAVE_TDM
    // Stage B with the Tensor Data Mover: one contiguous 1-row tile of
    // (ktiles*1024) 8-byte elements, global -> LDS offset 0 (Bsw).
    if (tid < 32) {                      // wave-uniform branch: single TDM issue
        unsigned long long ga = (unsigned long long)(size_t)Wsw;
        unsigned int nd = (unsigned int)(ktiles * 1024);   // u64 elements (<= 6144)
        v4u g0;
        g0[0] = 1u;                                        // count=1 (valid D#)
        g0[1] = 0u;                                        // lds_addr = 0 (Bsw base)
        g0[2] = (unsigned int)(ga & 0xffffffffull);        // global_addr[31:0]
        g0[3] = (unsigned int)((ga >> 32) & 0x1ffffffull)  // global_addr[56:32]
              | 0x80000000u;                               // type = 2 ("image")
        v8i g1;
        g1[0] = 0x00030000;                                // data_size = 8 bytes
        g1[1] = (int)((nd & 0xffffu) << 16);               // tensor_dim0 lo16
        g1[2] = (int)(((nd >> 16) & 0xffffu) | (1u << 16));// tensor_dim0 hi16 | tensor_dim1=1
        g1[3] = (int)((nd & 0xffffu) << 16);               // tile_dim0 = nd
        g1[4] = 1;                                         // tile_dim1 = 1, tile_dim2 = 0
        g1[5] = (int)nd;                                   // tensor_dim0_stride lo32
        g1[6] = 0;
        g1[7] = 0;
        v4i gz; gz[0] = 0; gz[1] = 0; gz[2] = 0; gz[3] = 0;
#if __has_include(<hip/amd_detail/amd_gfx1250_TDM.h>)
        v8i gz8;
        #pragma unroll
        for (int q = 0; q < 8; ++q) gz8[q] = 0;
        __builtin_amdgcn_tensor_load_to_lds(g0, g1, gz, gz, gz8, 0);   // 6-arg form
#else
        __builtin_amdgcn_tensor_load_to_lds(g0, g1, gz, gz, 0);        // 5-arg form
#endif
        __builtin_amdgcn_s_wait_tensorcnt(0);
    }
    __syncthreads();
    // The TDM write is invisible to the optimizer (the builtin carries no
    // pointer to Bsw): with no visible stores and a non-escaping address,
    // loads from Bsw legally fold to undef (observed in previous builds).
    // Escaping Bsw's address into a may-write asm defeats that proof while
    // leaving subsequent loads rooted at Bsw (keeps addrspace(3)/alignment,
    // so they still lower to ds_load_b128).
    {
        unsigned int* esc = &Bsw[0];
        asm volatile("" : "+v"(esc) : : "memory");
    }
#else
    {   // Fallback: cooperative contiguous copy.
        const int total = ktiles * 2048;
        for (int idx = tid; idx < total; idx += 256) Bsw[idx] = Wsw[idx];
        __syncthreads();
    }
#endif

    const long row0 = (long)blockIdx.x * 128 + (long)wave * 16;
    const int half = lane >> 4;
    const int nl   = lane & 15;

    // A row for this lane (branchless clamp keeps EXEC all-ones for WMMA).
    long mA = row0 + nl;
    mA = (mA < M) ? mA : 0;
    const unsigned short* Arow = A + mA * (long)Kpad;

    v8f acc[8];
    #pragma unroll
    for (int j = 0; j < 8; ++j)
        #pragma unroll
        for (int vi = 0; vi < 8; ++vi) acc[j][vi] = 0.0f;

    for (int kt = 0; kt < ktiles; ++kt) {
        union { v16bf v; unsigned int u[8]; } afrag;
        #pragma unroll
        for (int vi = 0; vi < 8; ++vi) {
            int k = (kt << 5) + ((vi >> 2) << 4) + (half << 3) + ((vi & 3) << 1);
            afrag.u[vi] = *(const unsigned int*)(Arow + k);   // 4B-aligned (k even)
        }
        #pragma unroll
        for (int j = 0; j < 8; ++j) {
            union { v16bf v; unsigned int u[8]; } bfrag;
            const unsigned int* p = &Bsw[(((kt * 8 + j) * 32) + lane) * 8];
            #pragma unroll
            for (int q = 0; q < 8; ++q) bfrag.u[q] = p[q];
            acc[j] = __builtin_amdgcn_wmma_f32_16x16x32_bf16(
                false, afrag.v, false, bfrag.v, (short)0, acc[j], false, false);
        }
    }

    // Epilogue. C/D layout: lane(half,nl), VGPR vi -> element (m = 8*half+vi, n = nl).
    #pragma unroll
    for (int j = 0; j < 8; ++j) {
        int n = (j << 4) + nl;
        float g  = gamma[n];
        float sh = bias[n] * g + beta[n];
        #pragma unroll
        for (int vi = 0; vi < 8; ++vi) {
            long mo = row0 + half * 8 + vi;
            if (mo < M) {
                float val = acc[j][vi] * g + sh;
                if (do_relu) val = fmaxf(val, 0.0f);
                long o = mo * H_DIM + n;
                if (residual) val += residual[o];
                if (outF) outF[o] = val;
                if (outB) outB[o] = f2bf(val);
            }
        }
    }
}

// -------------------------- edge scatter-add --------------------------------
// One wave per edge: 32 lanes x float4 = 128 features.
__global__ void scatter_kernel(const float* __restrict__ h,
                               const int* __restrict__ src,
                               const int* __restrict__ dst,
                               float* __restrict__ agg, int E) {
    int gw   = (blockIdx.x * blockDim.x + threadIdx.x) >> 5;
    int lane = threadIdx.x & 31;
    int nw   = (gridDim.x * blockDim.x) >> 5;
    for (int e = gw; e < E; e += nw) {
        long s = src[e];
        long d = dst[e];
        const float4 v = *(const float4*)(h + s * H_DIM + lane * 4);
        float* ap = agg + d * H_DIM + lane * 4;
        atomicAdd(ap + 0, v.x);
        atomicAdd(ap + 1, v.y);
        atomicAdd(ap + 2, v.z);
        atomicAdd(ap + 3, v.w);
    }
}

// z = bf16((1+eps)*h + agg)
__global__ void zmake_kernel(const float* __restrict__ h,
                             const float* __restrict__ agg,
                             const float* __restrict__ eps_g, int layer,
                             unsigned short* __restrict__ z, long total) {
    float e1 = 1.0f + eps_g[layer];
    long stride = (long)gridDim.x * blockDim.x;
    for (long i = (long)blockIdx.x * blockDim.x + threadIdx.x; i < total; i += stride)
        z[i] = f2bf(e1 * h[i] + agg[i]);
}

// ------------------------------ mean pool ----------------------------------
__global__ void pool_kernel(const float* __restrict__ h,
                            float* __restrict__ pooled, long R) {
    int c = threadIdx.x;   // 128 threads, one per feature
    float s = 0.0f;
    for (long r = blockIdx.x; r < R; r += gridDim.x) s += h[r * H_DIM + c];
    atomicAdd(&pooled[c], s);
}

// ------------------------------ classifier ---------------------------------
__global__ void classify_kernel(const float* __restrict__ pooled,
                                const float* __restrict__ Wc1,
                                const float* __restrict__ bc1,
                                const float* __restrict__ gc,
                                const float* __restrict__ bcn,
                                const float* __restrict__ Wc2,
                                const float* __restrict__ bc2,
                                float invN, float* __restrict__ out) {
    __shared__ float c[H2];
    int j = threadIdx.x;
    if (j < H2) {
        float s = 0.0f;
        for (int k = 0; k < H_DIM; ++k)
            s += pooled[k] * invN * Wc1[k * H2 + j];
        s += bc1[j];
        s = s * gc[j] + bcn[j];
        c[j] = fmaxf(s, 0.0f);
    }
    __syncthreads();
    if (j < NCLS) {
        float s = 0.0f;
        for (int q = 0; q < H2; ++q) s += c[q] * Wc2[q * NCLS + j];
        out[j] = s + bc2[j];
    }
}

// ---------------------------------------------------------------------------
extern "C" void kernel_launch(void* const* d_in, const int* in_sizes, int n_in,
                              void* d_out, int out_size, void* d_ws, size_t ws_size,
                              hipStream_t stream) {
    const float* x     = (const float*)d_in[0];
    const float* W_emb = (const float*)d_in[1];
    const float* b_emb = (const float*)d_in[2];
    const float* g_emb = (const float*)d_in[3];
    const float* be_emb= (const float*)d_in[4];
    const float* eps_g = (const float*)d_in[5];
    const float* W1    = (const float*)d_in[6];
    const float* b1    = (const float*)d_in[7];
    const float* ga1   = (const float*)d_in[8];
    const float* bb1   = (const float*)d_in[9];
    const float* W2    = (const float*)d_in[10];
    const float* b2    = (const float*)d_in[11];
    const float* ga2   = (const float*)d_in[12];
    const float* bb2   = (const float*)d_in[13];
    const float* Wc1   = (const float*)d_in[14];
    const float* bc1   = (const float*)d_in[15];
    const float* gc    = (const float*)d_in[16];
    const float* bcn   = (const float*)d_in[17];
    const float* Wc2   = (const float*)d_in[18];
    const float* bc2   = (const float*)d_in[19];
    const int*   ei    = (const int*)d_in[20];

    const long Nn = in_sizes[0] / F_IN;         // 100000
    const int  E  = in_sizes[20] / 2;           // 1600000
    const int  L  = in_sizes[5];                // 3

    // ---- workspace carve-out ----
    char* ws = (char*)d_ws;
    size_t off = 0;
    auto carve = [&](size_t bytes) -> char* {
        char* p = ws + off;
        off = (off + bytes + 255) & ~(size_t)255;
        return p;
    };
    unsigned short* xbf  = (unsigned short*)carve((size_t)Nn * F_IN_PAD * 2);
    float*          h    = (float*)carve((size_t)Nn * H_DIM * 4);
    float*          agg  = (float*)carve((size_t)Nn * H_DIM * 4);
    unsigned short* zbf  = (unsigned short*)carve((size_t)Nn * H_DIM * 2);
    unsigned short* tbf  = (unsigned short*)carve((size_t)Nn * H_DIM * 2);
    unsigned int*   wsw  = (unsigned int*)carve((size_t)6 * 2048 * 4);  // swizzled W
    float*          pooled = (float*)carve(H_DIM * 4);
    (void)ws_size;

    const long NH = Nn * H_DIM;
    const int gemm_blocks = (int)((Nn + 127) / 128);
    const int KT_EMB = F_IN_PAD / 32;   // 6
    const int KT_H   = H_DIM / 32;      // 4

    // 1) x -> bf16, padded to K=192
    pad_cols_kernel<<<2048, 256, 0, stream>>>(x, xbf, Nn, F_IN, F_IN_PAD);
    // 2) W_emb -> bf16 B-fragment order (zero-padded rows 165..191)
    swizzleW_kernel<<<48, 256, 0, stream>>>(W_emb, wsw, F_IN, KT_EMB);
    // 3) h = relu(bn(x @ W_emb + b_emb))
    gemm_bf16_wmma_kernel<<<gemm_blocks, 256, 0, stream>>>(
        xbf, wsw, Nn, F_IN_PAD, g_emb, b_emb, be_emb,
        h, (unsigned short*)nullptr, (const float*)nullptr, 1);

    for (int i = 0; i < L; ++i) {
        // agg = segment_sum(h[src], dst)
        fill_f32_kernel<<<2048, 256, 0, stream>>>(agg, NH, 0.0f);
        scatter_kernel<<<2048, 256, 0, stream>>>(h, ei, ei + E, agg, E);
        // z = (1+eps)*h + agg  (bf16)
        zmake_kernel<<<2048, 256, 0, stream>>>(h, agg, eps_g, i, zbf, NH);
        // t = relu(bn(z @ W1[i] + b1[i]))  (bf16)
        swizzleW_kernel<<<32, 256, 0, stream>>>(W1 + (size_t)i * H_DIM * H_DIM,
                                                wsw, H_DIM, KT_H);
        gemm_bf16_wmma_kernel<<<gemm_blocks, 256, 0, stream>>>(
            zbf, wsw, Nn, H_DIM,
            ga1 + i * H_DIM, b1 + i * H_DIM, bb1 + i * H_DIM,
            (float*)nullptr, tbf, (const float*)nullptr, 1);
        // h = bn(t @ W2[i] + b2[i]) + h  (residual, f32)
        swizzleW_kernel<<<32, 256, 0, stream>>>(W2 + (size_t)i * H_DIM * H_DIM,
                                                wsw, H_DIM, KT_H);
        gemm_bf16_wmma_kernel<<<gemm_blocks, 256, 0, stream>>>(
            tbf, wsw, Nn, H_DIM,
            ga2 + i * H_DIM, b2 + i * H_DIM, bb2 + i * H_DIM,
            h, (unsigned short*)nullptr, h, 0);
    }

    // mean pool + classifier head
    fill_f32_kernel<<<1, 128, 0, stream>>>(pooled, H_DIM, 0.0f);
    pool_kernel<<<512, 128, 0, stream>>>(h, pooled, Nn);
    classify_kernel<<<1, 64, 0, stream>>>(pooled, Wc1, bc1, gc, bcn, Wc2, bc2,
                                          1.0f / (float)Nn, (float*)d_out);
    (void)n_in; (void)out_size;
}
